// FixedOrderSlotAttention_57810259804265
// MI455X (gfx1250) — matrix operand order repaired and checked by reading, not compiled, not run on hardware
//
#include <hip/hip_runtime.h>
#include <hip/hip_bf16.h>

// ---------------------------------------------------------------------------
// FixedOrderSlotAttention for MI455X (gfx1250, wave32, WMMA bf16,
// async global->LDS staging)
// B=16, H=W=128, T=16384, D=64, S=8, MLP=128, 3 iterations
// ---------------------------------------------------------------------------

#define NB   16
#define NT   16384
#define ND   64
#define NW   128
#define NS   8

typedef __attribute__((ext_vector_type(16))) __bf16 v16bf;
typedef __attribute__((ext_vector_type(8)))  __bf16 v8bf;
typedef __attribute__((ext_vector_type(8)))  float  v8f;

static __device__ inline v8bf ld8(const __bf16* p) { return *(const v8bf*)p; }

static __device__ inline v16bf mk16(v8bf lo, v8bf hi) {
    v16bf r;
#pragma unroll
    for (int i = 0; i < 8; ++i) { r[i] = lo[i]; r[i + 8] = hi[i]; }
    return r;
}

static __device__ inline v16bf zero16() {
    v16bf r;
#pragma unroll
    for (int i = 0; i < 16; ++i) r[i] = (__bf16)0.0f;
    return r;
}

static __device__ inline v8f wmma_bf16(v16bf a, v16bf b, v8f c) {
    // D(f32 16x16) = A(bf16 16x32) * B(bf16 32x16) + C
    return __builtin_amdgcn_wmma_f32_16x16x32_bf16(
        false, a, false, b, (short)0, c, false, false);
}

// B-matrix (32x16, bf16) element for lane l, vector element e:
//   K = (l>>4)*16 + e, N = (l&15)   (contiguous 16 bf16 along K for fixed N)
// A-matrix (16x32, bf16) element for lane l, element e:
//   M = l&15;  K = (l>>4)*16 + e   (same contiguous K order as B)

// ---------------------------------------------------------------------------
// Small helper kernels
// ---------------------------------------------------------------------------
__global__ void convert_w_kernel(const float* __restrict__ wk,
                                 const float* __restrict__ wv,
                                 __bf16* __restrict__ wkb,
                                 __bf16* __restrict__ wvb) {
    int i = blockIdx.x * 256 + threadIdx.x;
    if (i < ND * ND) {
        wkb[i] = (__bf16)wk[i];
        wvb[i] = (__bf16)wv[i];
    }
}

__global__ void init_slots_kernel(const float* __restrict__ mu,
                                  const float* __restrict__ lsig,
                                  const float* __restrict__ noise,
                                  float* __restrict__ slots) {
    int i = blockIdx.x * 256 + threadIdx.x;
    if (i < NB * NS * ND) {
        int j = i & (NS * ND - 1);  // broadcast (1,S,D) over batch
        slots[i] = mu[j] + __expf(lsig[j]) * noise[i];
    }
}

__global__ void zero_iter_kernel(float* __restrict__ rowsum,
                                 float* __restrict__ updates) {
    int i = blockIdx.x * 256 + threadIdx.x;
    if (i < NB * NS) rowsum[i] = 0.0f;
    if (i < NB * NS * ND) updates[i] = 0.0f;
}

// ---------------------------------------------------------------------------
// k = x @ wk^T, v = x @ wv^T with x = LN(relu(enc)+pos) computed on the fly.
// One wave handles one 16-row tile of (B*T) rows; 4 waves / block.
// ---------------------------------------------------------------------------
__global__ __launch_bounds__(128) void kv_gemm_kernel(
    const float* __restrict__ enc, const float* __restrict__ pos_w,
    const float* __restrict__ pos_b, const float* __restrict__ lng,
    const float* __restrict__ lnb, const __bf16* __restrict__ wkb,
    const __bf16* __restrict__ wvb, __bf16* __restrict__ kb,
    __bf16* __restrict__ vb) {
    __shared__ alignas(16) __bf16 xs[4][16][ND];

    const int tid = threadIdx.x;
    const int wid = tid >> 5, lane = tid & 31;
    const int half = lane >> 4, l15 = lane & 15;
    const long tile = (long)blockIdx.x * 4 + wid;  // 16384 tiles total
    const long row0 = tile * 16;

    // Lanes 0..15: each builds one normalized bf16 row of x into LDS.
    if (half == 0) {
        const long row = row0 + l15;
        const int t = (int)(row & (NT - 1));
        const int i = t / NW, j = t & (NW - 1);
        const float e0 = 1.0f - i * (1.0f / 127.0f);
        const float e1 = i * (1.0f / 127.0f);
        const float e2 = j * (1.0f / 127.0f);
        const float e3 = 1.0f - j * (1.0f / 127.0f);
        const float* er = enc + row * ND;
        float y[ND];
        float m = 0.0f;
        for (int d = 0; d < ND; ++d) {
            float v = er[d];
            v = v > 0.0f ? v : 0.0f;
            v += pos_b[d] + e0 * pos_w[d * 4 + 0] + e1 * pos_w[d * 4 + 1] +
                 e2 * pos_w[d * 4 + 2] + e3 * pos_w[d * 4 + 3];
            y[d] = v;
            m += v;
        }
        m *= (1.0f / 64.0f);
        float var = 0.0f;
        for (int d = 0; d < ND; ++d) {
            float dv = y[d] - m;
            var += dv * dv;
        }
        var *= (1.0f / 64.0f);
        const float rs = rsqrtf(var + 1e-5f);
        for (int d = 0; d < ND; ++d)
            xs[wid][l15][d] = (__bf16)((y[d] - m) * rs * lng[d] + lnb[d]);
    }
    __syncthreads();

    // A tiles (16x32 bf16) from LDS, two K-chunks covering K=0..63.
    const __bf16* xr = &xs[wid][l15][0];
    const v16bf a0 = mk16(ld8(xr + half * 16), ld8(xr + half * 16 + 8));
    const v16bf a1 = mk16(ld8(xr + 32 + half * 16), ld8(xr + 32 + half * 16 + 8));

#pragma unroll
    for (int mat = 0; mat < 2; ++mat) {
        const __bf16* wb = mat ? wvb : wkb;
        __bf16* ob = mat ? vb : kb;
#pragma unroll
        for (int nt = 0; nt < 4; ++nt) {
            const int n = nt * 16 + l15;
            const __bf16* wr0 = wb + n * ND + half * 16;         // K chunk 0
            const __bf16* wr1 = wb + n * ND + 32 + half * 16;    // K chunk 1
            const v16bf b0 = mk16(ld8(wr0), ld8(wr0 + 8));
            const v16bf b1 = mk16(ld8(wr1), ld8(wr1 + 8));
            v8f acc = {};
            acc = wmma_bf16(a0, b0, acc);
            acc = wmma_bf16(a1, b1, acc);
#pragma unroll
            for (int r = 0; r < 8; ++r) {
                const long row = row0 + r + 8 * half;
                ob[row * ND + n] = (__bf16)acc[r];
            }
        }
    }
}

// ---------------------------------------------------------------------------
// q = LN(slots) @ wq^T * D^-0.5   -> bf16, padded to 16 slot rows (zeros)
// One 64-thread block per (b, padded-slot) row.
// ---------------------------------------------------------------------------
__global__ __launch_bounds__(64) void q_kernel(
    const float* __restrict__ slots, const float* __restrict__ lng,
    const float* __restrict__ lnb, const float* __restrict__ wq,
    __bf16* __restrict__ qb) {
    const int b = blockIdx.x >> 4, s = blockIdx.x & 15, d = threadIdx.x;
    __bf16* out = qb + ((long)b * 16 + s) * ND;
    if (s >= NS) {
        out[d] = (__bf16)0.0f;  // zero pad rows so WMMA A has clean zeros
        return;
    }
    __shared__ float sv[ND], red[ND];
    const float x = slots[((long)b * NS + s) * ND + d];
    red[d] = x;
    __syncthreads();
    for (int off = 32; off > 0; off >>= 1) {
        if (d < off) red[d] += red[d + off];
        __syncthreads();
    }
    const float m = red[0] * (1.0f / 64.0f);
    __syncthreads();
    const float dv = x - m;
    red[d] = dv * dv;
    __syncthreads();
    for (int off = 32; off > 0; off >>= 1) {
        if (d < off) red[d] += red[d + off];
        __syncthreads();
    }
    const float var = red[0] * (1.0f / 64.0f);
    __syncthreads();
    sv[d] = dv * rsqrtf(var + 1e-5f) * lng[d] + lnb[d];
    __syncthreads();
    float acc = 0.0f;
    const float* wr = wq + d * ND;
    for (int e = 0; e < ND; ++e) acc += wr[e] * sv[e];
    out[d] = (__bf16)(acc * 0.125f);  // 64^-0.5
}

// ---------------------------------------------------------------------------
// dots = q @ k^T (WMMA), softmax over slots per key column, store attn
// numerators (bf16, (s,t) layout) and atomically accumulate per-slot key sums.
// One wave per 16-key tile.
// ---------------------------------------------------------------------------
__global__ __launch_bounds__(128) void dots_kernel(
    const __bf16* __restrict__ qb, const __bf16* __restrict__ kb,
    __bf16* __restrict__ attn, float* __restrict__ rowsum) {
    const int tid = threadIdx.x;
    const int wid = tid >> 5, lane = tid & 31;
    const int half = lane >> 4, l15 = lane & 15;
    const int gw = blockIdx.x * 4 + wid;   // 16384 waves
    const int b = gw >> 10;                // 1024 key tiles per batch
    const int t0 = (gw & 1023) * 16;

    // A: q rows (slot = l15; rows >= 8 are zero-filled in the q buffer)
    const __bf16* qr = qb + ((long)b * 16 + l15) * ND;
    const v16bf a0 = mk16(ld8(qr + half * 16), ld8(qr + half * 16 + 8));
    const v16bf a1 = mk16(ld8(qr + 32 + half * 16), ld8(qr + 32 + half * 16 + 8));

    // B: k rows; contracts over D so rows are contiguous.
    const __bf16* kr = kb + ((long)b * NT + t0 + l15) * ND;
    const v16bf b0 = mk16(ld8(kr + half * 16), ld8(kr + half * 16 + 8));
    const v16bf b1 = mk16(ld8(kr + 32 + half * 16), ld8(kr + 32 + half * 16 + 8));

    v8f acc = {};
    acc = wmma_bf16(a0, b0, acc);
    acc = wmma_bf16(a1, b1, acc);

    // Column softmax over the 8 real slots. In the 16x16 f32 D layout, lane n
    // (half==0) holds M=0..7 of column n in acc[0..7] -> entirely in-lane.
    float p[8];
#pragma unroll
    for (int r = 0; r < 8; ++r) p[r] = 0.0f;
    if (half == 0) {
        float mx = acc[0];
#pragma unroll
        for (int r = 1; r < 8; ++r) mx = fmaxf(mx, acc[r]);
        float ssum = 0.0f;
#pragma unroll
        for (int r = 0; r < 8; ++r) {
            p[r] = __expf(acc[r] - mx);
            ssum += p[r];
        }
        const float inv = 1.0f / ssum;
        const long tcol = (long)t0 + l15;
#pragma unroll
        for (int r = 0; r < 8; ++r) {
            p[r] = p[r] * inv + 1e-8f;
            attn[((long)b * NS + r) * NT + tcol] = (__bf16)p[r];
        }
    }
    // Per-slot partial key sums: reduce across the 16-lane group, one atomic.
#pragma unroll
    for (int r = 0; r < 8; ++r) {
        float v = p[r];
        v += __shfl_xor(v, 1, 32);
        v += __shfl_xor(v, 2, 32);
        v += __shfl_xor(v, 4, 32);
        v += __shfl_xor(v, 8, 32);
        if (lane == 0) atomicAdd(&rowsum[b * NS + r], v);
    }
}

// ---------------------------------------------------------------------------
// updates = attn @ v (WMMA over K = T). v tiles are async-copied straight
// from global memory into this wave's private LDS quadrant
// (GLOBAL_LOAD_ASYNC_TO_LDS_B128, tracked by ASYNCcnt -> no VGPR round-trip,
// no block barrier needed), then gathered transposed for the B operand.
// 8 waves / batch, partials combined with global f32 atomics.
// ---------------------------------------------------------------------------
__global__ __launch_bounds__(128) void updates_kernel(
    const __bf16* __restrict__ attn, const __bf16* __restrict__ vb,
    float* __restrict__ updates) {
    __shared__ alignas(16) __bf16 vs[4][32][ND];

    const int tid = threadIdx.x;
    const int wid = tid >> 5, lane = tid & 31;
    const int half = lane >> 4, l15 = lane & 15;
    const int gw = blockIdx.x * 4 + wid;   // 128 waves
    const int b = gw >> 3;
    const long tbase = (long)(gw & 7) * 2048;

    // Wave-relative LDS byte offset of this lane's destination row: on
    // AMDGPU the low 32 bits of a flat pointer into LDS are the LDS offset.
    const unsigned ldsrow =
        (unsigned)(unsigned long)(&vs[wid][lane][0]);

    v8f acc[4];
#pragma unroll
    for (int nt = 0; nt < 4; ++nt) acc[nt] = (v8f){};

    for (int step = 0; step < 64; ++step) {
        const long t0 = tbase + step * 32;
        // Async-copy one 32(t) x 64(d) bf16 v tile (128B row per lane)
        // directly into LDS. IOFFSET applies to both LDS and global sides.
        const __bf16* gsrc = vb + ((long)b * NT + t0 + lane) * ND;
        asm volatile(
            "global_load_async_to_lds_b128 %0, %1, off\n\t"
            "global_load_async_to_lds_b128 %0, %1, off offset:16\n\t"
            "global_load_async_to_lds_b128 %0, %1, off offset:32\n\t"
            "global_load_async_to_lds_b128 %0, %1, off offset:48\n\t"
            "global_load_async_to_lds_b128 %0, %1, off offset:64\n\t"
            "global_load_async_to_lds_b128 %0, %1, off offset:80\n\t"
            "global_load_async_to_lds_b128 %0, %1, off offset:96\n\t"
            "global_load_async_to_lds_b128 %0, %1, off offset:112"
            :
            : "v"(ldsrow), "v"(gsrc)
            : "memory");

        // A: attn numerators, row M = slot (zeros for padded slots 8..15).
        v16bf a = zero16();
        if (l15 < NS) {
            const __bf16* ar = attn + ((long)b * NS + l15) * NT + t0;
            a = mk16(ld8(ar + half * 16), ld8(ar + half * 16 + 8));
        }
        if (step < 63)
            __builtin_prefetch(vb + ((long)b * NT + t0 + 32 + lane) * ND, 0, 0);

        // Wait for this wave's async tile before the transposed gathers.
        asm volatile("s_wait_asynccnt 0x0" ::: "memory");

#pragma unroll
        for (int nt = 0; nt < 4; ++nt) {
            const int n = nt * 16 + l15;
            v16bf bbv;
#pragma unroll
            for (int e = 0; e < 16; ++e)
                bbv[e] = vs[wid][half * 16 + e][n];  // transposed LDS gather
            acc[nt] = wmma_bf16(a, bbv, acc[nt]);
        }
        // Program order guarantees the ds gathers above have drained into
        // VGPRs (WMMA consumed them) before the next step's async writes.
    }

    if (half == 0) {  // rows M=0..7 are the real slots
#pragma unroll
        for (int nt = 0; nt < 4; ++nt)
#pragma unroll
            for (int r = 0; r < 8; ++r)
                atomicAdd(&updates[((long)b * NS + r) * ND + nt * 16 + l15],
                          acc[nt][r]);
    }
}

// ---------------------------------------------------------------------------
// GRU cell + LN + MLP residual for the 128 slot rows (plain f32 VALU).
// One 64-thread block per (b, s) row.
// ---------------------------------------------------------------------------
__global__ __launch_bounds__(64) void gru_mlp_kernel(
    const float* __restrict__ updates, const float* __restrict__ rowsum,
    const float* __restrict__ slots_in, const float* __restrict__ wih,
    const float* __restrict__ whh, const float* __restrict__ bih,
    const float* __restrict__ bhh, const float* __restrict__ lng,
    const float* __restrict__ lnb, const float* __restrict__ w1,
    const float* __restrict__ b1, const float* __restrict__ w2,
    const float* __restrict__ b2, float* __restrict__ slots_out) {
    const int b = blockIdx.x >> 3, s = blockIdx.x & 7, d = threadIdx.x;
    const long base = ((long)b * NS + s) * ND;
    __shared__ float u[ND], h[ND], sm[ND], hid[128], red[ND];

    const float rs_inv = 1.0f / rowsum[b * NS + s];
    u[d] = updates[base + d] * rs_inv;
    const float hv = slots_in[base + d];
    h[d] = hv;
    __syncthreads();

    float ir = bih[d], iz = bih[ND + d], in = bih[2 * ND + d];
    float hr = bhh[d], hz = bhh[ND + d], hn = bhh[2 * ND + d];
    for (int e = 0; e < ND; ++e) {
        const float ue = u[e], he = h[e];
        ir += wih[d * ND + e] * ue;
        iz += wih[(ND + d) * ND + e] * ue;
        in += wih[(2 * ND + d) * ND + e] * ue;
        hr += whh[d * ND + e] * he;
        hz += whh[(ND + d) * ND + e] * he;
        hn += whh[(2 * ND + d) * ND + e] * he;
    }
    const float r = 1.0f / (1.0f + __expf(-(ir + hr)));
    const float z = 1.0f / (1.0f + __expf(-(iz + hz)));
    const float n = tanhf(in + r * hn);
    const float hnew = (1.0f - z) * n + z * hv;

    red[d] = hnew;
    __syncthreads();
    for (int off = 32; off > 0; off >>= 1) {
        if (d < off) red[d] += red[d + off];
        __syncthreads();
    }
    const float m = red[0] * (1.0f / 64.0f);
    __syncthreads();
    const float dv = hnew - m;
    red[d] = dv * dv;
    __syncthreads();
    for (int off = 32; off > 0; off >>= 1) {
        if (d < off) red[d] += red[d + off];
        __syncthreads();
    }
    const float var = red[0] * (1.0f / 64.0f);
    __syncthreads();
    sm[d] = dv * rsqrtf(var + 1e-5f) * lng[d] + lnb[d];
    __syncthreads();

#pragma unroll
    for (int jj = 0; jj < 2; ++jj) {
        const int jh = d + jj * ND;
        float a = b1[jh];
        for (int e = 0; e < ND; ++e) a += w1[jh * ND + e] * sm[e];
        hid[jh] = fmaxf(a, 0.0f);
    }
    __syncthreads();
    float o = b2[d];
    for (int e = 0; e < 128; ++e) o += w2[d * 128 + e] * hid[e];
    slots_out[base + d] = hnew + o;
}

// ---------------------------------------------------------------------------
// Host launch
// ---------------------------------------------------------------------------
extern "C" void kernel_launch(void* const* d_in, const int* in_sizes, int n_in,
                              void* d_out, int out_size, void* d_ws,
                              size_t ws_size, hipStream_t stream) {
    const float* enc       = (const float*)d_in[0];
    const float* pos_w     = (const float*)d_in[1];
    const float* pos_b     = (const float*)d_in[2];
    const float* ln_in_g   = (const float*)d_in[3];
    const float* ln_in_b   = (const float*)d_in[4];
    const float* ln_slot_g = (const float*)d_in[5];
    const float* ln_slot_b = (const float*)d_in[6];
    const float* ln_mlp_g  = (const float*)d_in[7];
    const float* ln_mlp_b  = (const float*)d_in[8];
    const float* wq        = (const float*)d_in[9];
    const float* wk        = (const float*)d_in[10];
    const float* wv        = (const float*)d_in[11];
    const float* gru_wih   = (const float*)d_in[12];
    const float* gru_whh   = (const float*)d_in[13];
    const float* gru_bih   = (const float*)d_in[14];
    const float* gru_bhh   = (const float*)d_in[15];
    const float* mlp_w1    = (const float*)d_in[16];
    const float* mlp_b1    = (const float*)d_in[17];
    const float* mlp_w2    = (const float*)d_in[18];
    const float* mlp_b2    = (const float*)d_in[19];
    const float* mu        = (const float*)d_in[20];
    const float* lsig      = (const float*)d_in[21];
    const float* noise     = (const float*)d_in[22];

    char* ws = (char*)d_ws;
    __bf16* kb    = (__bf16*)(ws);                       // 32 MB
    __bf16* vb    = (__bf16*)(ws + 33554432);            // 32 MB
    __bf16* attn  = (__bf16*)(ws + 67108864);            // 4 MB
    __bf16* qb    = (__bf16*)(ws + 71303168);            // 32 KB
    float*  slots = (float*)(ws + 71335936);             // 32 KB
    float*  rowsum= (float*)(ws + 71368704);             // 1 KB
    float*  upd   = (float*)(ws + 71369728);             // 32 KB
    __bf16* wkb   = (__bf16*)(ws + 71402496);            // 8 KB
    __bf16* wvb   = (__bf16*)(ws + 71410688);            // 8 KB

    convert_w_kernel<<<16, 256, 0, stream>>>(wk, wv, wkb, wvb);
    kv_gemm_kernel<<<4096, 128, 0, stream>>>(enc, pos_w, pos_b, ln_in_g,
                                             ln_in_b, wkb, wvb, kb, vb);
    init_slots_kernel<<<32, 256, 0, stream>>>(mu, lsig, noise, slots);

    for (int it = 0; it < 3; ++it) {
        zero_iter_kernel<<<32, 256, 0, stream>>>(rowsum, upd);
        q_kernel<<<256, 64, 0, stream>>>(slots, ln_slot_g, ln_slot_b, wq, qb);
        dots_kernel<<<4096, 128, 0, stream>>>(qb, kb, attn, rowsum);
        updates_kernel<<<32, 128, 0, stream>>>(attn, vb, upd);
        float* so = (it == 2) ? (float*)d_out : slots;
        gru_mlp_kernel<<<128, 64, 0, stream>>>(
            upd, rowsum, slots, gru_wih, gru_whh, gru_bih, gru_bhh, ln_mlp_g,
            ln_mlp_b, mlp_w1, mlp_b1, mlp_w2, mlp_b2, so);
    }
}
